// LSTM_41764261986630
// MI455X (gfx1250) — matrix operand order repaired and checked by reading, not compiled, not run on hardware
//
#include <hip/hip_runtime.h>

// ---------------------------------------------------------------------------
// 2-layer LSTM (T=512, B=64, IN=H=512) for MI455X / gfx1250.
//
// Persistent kernel, bf16 WMMA (f32 accumulate), per-WG weight slices
// resident in 320KB LDS for all 512 timesteps, cell state in registers,
// one grid sync per timestep. See prior round for roofline reasoning:
// ~275 GFLOP, L2-resident weights, recurrence-latency bound.
// ---------------------------------------------------------------------------

#define TSTEPS 512
#define NB     64
#define NIN    512
#define NH     512
#define NWG    64          // 32 h'-tiles x 2 batch halves
#define NTHR   256         // 8 wave32 = 4 gates x 2 m-tiles

typedef __attribute__((ext_vector_type(16))) __bf16   v16bf;
typedef __attribute__((ext_vector_type(8)))  float    v8f;
typedef __attribute__((ext_vector_type(4)))  unsigned u32x4;

union ABPack { u32x4 u[2]; v16bf v; };

__device__ __forceinline__ unsigned short f2bf(float f) {
  unsigned u = __float_as_uint(f);
  u += 0x7FFFu + ((u >> 16) & 1u);       // round-to-nearest-even
  return (unsigned short)(u >> 16);
}

__device__ __forceinline__ float sigmoidf_(float x) {
  return 1.0f / (1.0f + __expf(-x));
}

// ---------------- workspace layout (bytes) ----------------
//  [0,256)          : grid-sync {cnt, gen}
//  [WS_XB, +32MB)   : x as bf16, [T][B][IN] row-major
//  [WS_WT, +8MB)    : weights pre-swizzled into WMMA-B layout:
//                     [layer2][jtile128][kstep32][lane32][n16] bf16
//  [WS_ACT, +256KB) : h activations bf16, 4 buffers [layer][parity][B][H]
#define WS_XB   256
#define WS_WT   (WS_XB + TSTEPS * NB * NIN * 2)
#define WS_ACT  (WS_WT + 2 * (4 * NH) * (NIN + NH) * 2)

// LDS layout inside persistent kernel
#define LDS_GAT_OFF   262144          // after 8 slots x 32KB of weights
#define LDS_B_OFF     270336          // 128 bias floats
#define LDS_TOTAL     270848

// activation buffer stride (halves): one [B][H] bf16 buffer
#define ACT_STRIDE    (NB * NH)

// ---------------------------------------------------------------------------
__global__ void k_init(unsigned* sync, unsigned short* act) {
  int i = blockIdx.x * blockDim.x + threadIdx.x;
  if (i < 8) sync[i] = 0u;
  for (int j = i; j < 4 * NB * NH; j += gridDim.x * blockDim.x) act[j] = 0;
}

__global__ void k_cvt_x(const float* __restrict__ x, unsigned short* __restrict__ xb) {
  int i = blockIdx.x * 256 + threadIdx.x;             // grid sized exactly
  xb[i] = f2bf(x[i]);
}

// Scatter Wcat = [Wx | Wh] into the exact WMMA B-operand order so the hot
// loop reads one contiguous 32B chunk per lane per k-step.
// B (32x16 bf16): lane L holds column N=L%16; lanes<16 take K {0..7,16..23},
//                 lanes>=16 take K {8..15,24..31} (same order as A).
__global__ void k_cvt_w(const float* __restrict__ Wx, const float* __restrict__ Wh,
                        unsigned short* __restrict__ wt) {
  int i  = blockIdx.x * 256 + threadIdx.x;  // [l][jt][ks][ln][n]
  int n  = i & 15;
  int ln = (i >> 4) & 31;
  int ks = (i >> 9) & 31;
  int jt = (i >> 14) & 127;
  int l  = (i >> 21) & 1;
  int j = jt * 16 + (ln & 15);
  int k = ks * 32 + 8 * (ln >> 4) + (n & 7) + 16 * (n >> 3);
  float w = (k < NIN) ? Wx[(l * 4 * NH + j) * NIN + k]
                      : Wh[(l * 4 * NH + j) * NH + (k - NIN)];
  wt[i] = f2bf(w);
}

// ---------------------------------------------------------------------------
__device__ __forceinline__ void grid_sync(unsigned* cnt, unsigned* gen) {
  __threadfence();
  __syncthreads();
  if (threadIdx.x == 0) {
    unsigned g = __hip_atomic_load(gen, __ATOMIC_RELAXED, __HIP_MEMORY_SCOPE_AGENT);
    unsigned arrived =
        __hip_atomic_fetch_add(cnt, 1u, __ATOMIC_ACQ_REL, __HIP_MEMORY_SCOPE_AGENT) + 1u;
    if (arrived == NWG) {
      __hip_atomic_store(cnt, 0u, __ATOMIC_RELAXED, __HIP_MEMORY_SCOPE_AGENT);
      __hip_atomic_fetch_add(gen, 1u, __ATOMIC_RELEASE, __HIP_MEMORY_SCOPE_AGENT);
    } else {
      while (__hip_atomic_load(gen, __ATOMIC_ACQUIRE, __HIP_MEMORY_SCOPE_AGENT) == g)
        __builtin_amdgcn_s_sleep(8);
    }
  }
  __syncthreads();
  __threadfence();
}

// One 16x16 output tile, K = 1024 split as 16 k-steps from src1 + 16 from src2.
// A (16x32 bf16) per ISA layout: lane L handles row M=L%16; lanes<16 take
// K {0..7,16..23}, lanes>=16 take K {8..15,24..31}  ->  two b128 loads at
// rowbase + (L>=16?16:0) and +32.
__device__ __forceinline__ v8f mma_tile(const unsigned short* __restrict__ a1,
                                        const unsigned short* __restrict__ a2,
                                        const char* __restrict__ bsrc) {
  v8f c = {};
#pragma unroll
  for (int ks = 0; ks < 16; ++ks) {
    ABPack a;
    a.u[0] = *(const u32x4*)(a1 + ks * 32);
    a.u[1] = *(const u32x4*)(a1 + ks * 32 + 16);
    v16bf b = *(const v16bf*)(bsrc + ks * 1024);
    c = __builtin_amdgcn_wmma_f32_16x16x32_bf16(false, a.v, false, b, (short)0, c,
                                                false, false);
  }
#pragma unroll
  for (int ks = 0; ks < 16; ++ks) {
    ABPack a;
    a.u[0] = *(const u32x4*)(a2 + ks * 32);
    a.u[1] = *(const u32x4*)(a2 + ks * 32 + 16);
    v16bf b = *(const v16bf*)(bsrc + (16 + ks) * 1024);
    c = __builtin_amdgcn_wmma_f32_16x16x32_bf16(false, a.v, false, b, (short)0, c,
                                                false, false);
  }
  return c;
}

__device__ __forceinline__ void elem_phase(const float* gat, const float* blds_l,
                                           float* creg, unsigned short* hdst,
                                           float* outdst, float* houtdst,
                                           float* coutdst, bool last, int tid,
                                           int mb, int hp) {
#pragma unroll
  for (int e = 0; e < 2; ++e) {
    int idx = tid + e * 256;           // 512 (b,h') elements per WG
    int emtl = idx >> 8;
    int m = (idx >> 4) & 15;
    int n = idx & 15;
    int off = emtl * 256 + m * 16 + n;
    float iv = sigmoidf_(gat[0 * 512 + off] + blds_l[0 * 16 + n]);
    float fv = sigmoidf_(gat[1 * 512 + off] + blds_l[1 * 16 + n]);
    float gv = tanhf(gat[2 * 512 + off] + blds_l[2 * 16 + n]);
    float ov = sigmoidf_(gat[3 * 512 + off] + blds_l[3 * 16 + n]);
    float cc = fv * creg[e] + iv * gv;   // cell state lives in registers
    creg[e] = cc;
    float hh = ov * tanhf(cc);
    int b  = (mb * 2 + emtl) * 16 + m;
    int hq = hp * 16 + n;
    hdst[b * NH + hq] = f2bf(hh);
    if (outdst) outdst[b * NH + hq] = hh;
    if (last) { houtdst[b * NH + hq] = hh; coutdst[b * NH + hq] = cc; }
  }
}

__global__ void __launch_bounds__(NTHR, 1)
k_lstm(const unsigned short* __restrict__ xb, const unsigned short* __restrict__ wt,
       const float* __restrict__ bh, unsigned short* __restrict__ act,
       float* __restrict__ out, float* __restrict__ hout, float* __restrict__ cout,
       unsigned* __restrict__ sync) {
  extern __shared__ char smem[];
  unsigned short* wlds = (unsigned short*)smem;
  float* gat  = (float*)(smem + LDS_GAT_OFF);
  float* blds = (float*)(smem + LDS_B_OFF);

  const int tid = threadIdx.x;
  const int wg  = blockIdx.x;
  const int mb  = wg >> 5;        // batch half (rows 0..31 / 32..63)
  const int hp  = wg & 31;        // h'-tile (16 hidden units)
  const int ln  = tid & 31;
  const int wid = tid >> 5;
  const int gate = wid >> 1;      // i,f,g,o
  const int mtl  = wid & 1;
  const int mt   = mb * 2 + mtl;

  // Stage this WG's weight slices (both layers, all 4 gates) into LDS once.
  for (int l = 0; l < 2; ++l)
    for (int g = 0; g < 4; ++g) {
      const u32x4* srcw = (const u32x4*)(wt + (l * 128 + g * 32 + hp) * 16384);
      u32x4* dstw = (u32x4*)(wlds + (l * 4 + g) * 16384);
      for (int q = tid; q < 2048; q += NTHR) dstw[q] = srcw[q];
    }
  if (tid < 128) {
    int l = tid >> 6, g = (tid >> 4) & 3, n = tid & 15;
    blds[tid] = bh[l * 4 * NH + g * NH + hp * 16 + n];
  }
  __syncthreads();

  const int mrow = mt * 16 + (ln & 15);
  // per-lane half-offset into a row (8 halves for lanes>=16), see mma_tile
  const int aOff = (ln >= 16) ? 8 : 0;
  const int rowOff = mrow * NH + aOff;          // offset in halves within a buffer
  const char* bsrc0 = (const char*)(wlds + (0 * 4 + gate) * 16384) + ln * 32;
  const char* bsrc1 = (const char*)(wlds + (1 * 4 + gate) * 16384) + ln * 32;

  float creg0[2] = {0.f, 0.f};
  float creg1[2] = {0.f, 0.f};

  for (int t = 0; t < TSTEPS; ++t) {
    const int p = t & 1, pp = p ^ 1;
    const bool last = (t == TSTEPS - 1);
    const int rr = (ln >= 16) ? 8 : 0, nn = ln & 15;

    // Keep all activation pointers derived arithmetically from the global
    // kernel arg so addrspace inference yields global_load_b128 (not flat).
    // buffers (in halves): h0[p] = act + p*ACT_STRIDE, h1[p] = act + (2+p)*ACT_STRIDE
    // ---- layer 0: ifgo = [x_t | h0_prev] @ Wcat0^T ----
    {
      const unsigned short* a1 = xb + ((size_t)t * NB + mrow) * NIN + aOff;
      const unsigned short* a2 = act + (size_t)pp * ACT_STRIDE + rowOff;
      v8f c = mma_tile(a1, a2, bsrc0);
      float* gslot = gat + (gate * 2 + mtl) * 256;   // C/D layout -> LDS
#pragma unroll
      for (int r = 0; r < 8; ++r) gslot[(r + rr) * 16 + nn] = c[r];
    }
    __syncthreads();
    elem_phase(gat, blds, creg0, act + (size_t)p * ACT_STRIDE, nullptr,
               hout, cout, last, tid, mb, hp);

    grid_sync(sync, sync + 1);   // all h0[t] visible device-wide

    // ---- layer 1: ifgo = [h0_cur | h1_prev] @ Wcat1^T ----
    {
      const unsigned short* a1 = act + (size_t)p * ACT_STRIDE + rowOff;
      const unsigned short* a2 = act + (size_t)(2 + pp) * ACT_STRIDE + rowOff;
      v8f c = mma_tile(a1, a2, bsrc1);
      float* gslot = gat + (gate * 2 + mtl) * 256;
#pragma unroll
      for (int r = 0; r < 8; ++r) gslot[(r + rr) * 16 + nn] = c[r];
    }
    __syncthreads();
    elem_phase(gat, blds + 64, creg1, act + (size_t)(2 + p) * ACT_STRIDE,
               out + (size_t)t * NB * NH, hout + NB * NH, cout + NB * NH,
               last, tid, mb, hp);
    __syncthreads();
  }
}

// ---------------------------------------------------------------------------
extern "C" void kernel_launch(void* const* d_in, const int* in_sizes, int n_in,
                              void* d_out, int out_size, void* d_ws, size_t ws_size,
                              hipStream_t stream) {
  const float* x  = (const float*)d_in[0];
  const float* Wx = (const float*)d_in[1];
  const float* Wh = (const float*)d_in[2];
  const float* bh = (const float*)d_in[3];
  float* out = (float*)d_out;
  char* ws = (char*)d_ws;

  unsigned*       sync = (unsigned*)ws;
  unsigned short* xb   = (unsigned short*)(ws + WS_XB);
  unsigned short* wt   = (unsigned short*)(ws + WS_WT);
  unsigned short* act  = (unsigned short*)(ws + WS_ACT);

  (void)in_sizes; (void)n_in; (void)out_size; (void)ws_size;

  hipFuncSetAttribute((const void*)k_lstm,
                      hipFuncAttributeMaxDynamicSharedMemorySize, LDS_TOTAL);

  k_init<<<64, 256, 0, stream>>>(sync, act);
  k_cvt_x<<<(TSTEPS * NB * NIN) / 256, 256, 0, stream>>>(x, xb);
  k_cvt_w<<<(2 * 128 * 32 * 32 * 16) / 256, 256, 0, stream>>>(Wx, Wh, wt);

  float* hout = out + (size_t)TSTEPS * NB * NH;
  float* cout = hout + 2 * NB * NH;
  k_lstm<<<NWG, NTHR, LDS_TOTAL, stream>>>(xb, wt, bh, act, out, hout, cout, sync);
}
